// SimpleGNN_17257178596102
// MI455X (gfx1250) — matrix operand order repaired and checked by reading, not compiled, not run on hardware
//
#include <hip/hip_runtime.h>
#include <hip/hip_bf16.h>

typedef __attribute__((ext_vector_type(16))) _Float16 v16h;
typedef __attribute__((ext_vector_type(8)))  _Float16 v8h;
typedef __attribute__((ext_vector_type(8)))  float    v8f;

#define NEG_SLOPE 0.2f
#define EPSV 1e-16f

// ---------------- utility: ordered-int float max (monotone map) ----------------
__device__ __forceinline__ int orderF(float f) {
    int i = __float_as_int(f);
    return (i >= 0) ? i : (i ^ 0x7fffffff);
}
__device__ __forceinline__ float unorderF(int i) {
    return __int_as_float((i >= 0) ? i : (i ^ 0x7fffffff));
}

// ---------------- f32 -> f16 convert ----------------
__global__ __launch_bounds__(256) void k_cvt_f16(const float* __restrict__ x,
                                                 _Float16* __restrict__ y, long n) {
    long i = (long)blockIdx.x * blockDim.x + threadIdx.x;
    if (i < n) y[i] = (_Float16)x[i];
}

// ---------------- transpose W[K,C] -> Wt[C,K] as f16 ----------------
__global__ __launch_bounds__(256) void k_transpose_f16(const float* __restrict__ W,
                                                       _Float16* __restrict__ Wt,
                                                       int K, int C) {
    long t = (long)blockIdx.x * blockDim.x + threadIdx.x;
    if (t >= (long)K * C) return;
    int c = (int)(t / K);
    int k = (int)(t % K);
    Wt[t] = (_Float16)W[(long)k * C + c];  // Wt[c*K + k]
}

// ---------------- WMMA GEMM: Out[M,Nc] = A[M,K](f16) * Bt[Nc,K]^T ----------------
// One wave per 16x64 strip (4 tiles): A fragment loaded once, 4 WMMAs per K-step.
// K multiple of 32; M multiple of 16; Nc multiple of 64.
__global__ __launch_bounds__(256) void k_gemm_wmma(const _Float16* __restrict__ A,
                                                   const _Float16* __restrict__ Bt,
                                                   float* __restrict__ Out,
                                                   int M, int K, int Nc) {
    int wave = (int)((blockIdx.x * blockDim.x + threadIdx.x) >> 5);
    int lane = threadIdx.x & 31;
    int ngroups = Nc >> 6;                // groups of 4 column tiles
    int mt = wave / ngroups;
    int ng = wave % ngroups;
    if (mt * 16 >= M) return;  // wave-uniform: EXEC stays all-ones for active waves

    int hi  = lane >> 4;   // 0: lanes 0-15, 1: lanes 16-31
    int l15 = lane & 15;

    // A fragment (16x32 f16): lane<16 holds row l15, K {kb..kb+7, kb+16..kb+23};
    // lane>=16 holds row l15, K {kb+8..kb+15, kb+24..kb+31}  (ISA 7.12.2)
    const _Float16* arow = A  + (size_t)(mt * 16 + l15) * K + (hi ? 8 : 0);
    // B fragment (32x16 f16): lane<16 = col l15, K {kb..kb+7, kb+16..kb+23}; lane>=16 K {kb+8..kb+15, kb+24..kb+31}
    const _Float16* bbase = Bt + (size_t)(ng * 64 + l15) * K + (hi ? 8 : 0);
    const size_t bstride = (size_t)16 * K;  // next column tile

    v8f acc[4];
#pragma unroll
    for (int s = 0; s < 4; ++s) acc[s] = (v8f){0.f, 0.f, 0.f, 0.f, 0.f, 0.f, 0.f, 0.f};

    for (int kb = 0; kb < K; kb += 32) {
        v8h a0 = *(const v8h*)(arow + kb);
        v8h a1 = *(const v8h*)(arow + kb + 16);
        v16h a;
#pragma unroll
        for (int j = 0; j < 8; ++j) { a[j] = a0[j]; a[j + 8] = a1[j]; }
#pragma unroll
        for (int s = 0; s < 4; ++s) {
            const _Float16* brow = bbase + (size_t)s * bstride + kb;
            v8h b0 = *(const v8h*)(brow);
            v8h b1 = *(const v8h*)(brow + 16);
            v16h b;
#pragma unroll
            for (int j = 0; j < 8; ++j) { b[j] = b0[j]; b[j + 8] = b1[j]; }
            acc[s] = __builtin_amdgcn_wmma_f32_16x16x32_f16(false, a, false, b,
                                                            (short)0, acc[s], false, false);
        }
    }
    // D layout: VGPR i -> row (hi*8 + i), col = l15
#pragma unroll
    for (int s = 0; s < 4; ++s) {
        float* orow = Out + (size_t)(mt * 16 + hi * 8) * Nc + ng * 64 + s * 16 + l15;
#pragma unroll
        for (int i = 0; i < 8; ++i) orow[(size_t)i * Nc] = acc[s][i];
    }
}

// ---------------- per-node attention scores a_src/a_dst [N,4] ----------------
__global__ __launch_bounds__(128) void k_attn(const float* __restrict__ h,
                                              const float* __restrict__ att_s,
                                              const float* __restrict__ att_d,
                                              float* __restrict__ a_s,
                                              float* __restrict__ a_d, int N) {
    int n = blockIdx.x;
    if (n >= N) return;
    int hd   = threadIdx.x >> 5;  // head 0..3 (one wave per head)
    int lane = threadIdx.x & 31;
    const float* row = h + (size_t)n * 256 + hd * 64;
    float v0 = row[lane], v1 = row[lane + 32];
    float s = v0 * att_s[hd * 64 + lane] + v1 * att_s[hd * 64 + lane + 32];
    float d = v0 * att_d[hd * 64 + lane] + v1 * att_d[hd * 64 + lane + 32];
#pragma unroll
    for (int off = 16; off; off >>= 1) {
        s += __shfl_xor(s, off, 32);
        d += __shfl_xor(d, off, 32);
    }
    if (lane == 0) { a_s[n * 4 + hd] = s; a_d[n * 4 + hd] = d; }
}

// ---------------- init: agg = bias broadcast; nmax = -inf; denom = 0 ----------------
__global__ __launch_bounds__(256) void k_init_agg(const float* __restrict__ bias,
                                                  float* __restrict__ agg, long n) {
    long i = (long)blockIdx.x * blockDim.x + threadIdx.x;
    if (i < n) agg[i] = bias[i & 255];
}
__global__ __launch_bounds__(256) void k_init_softmax(int* __restrict__ nmax,
                                                      float* __restrict__ denom, long n) {
    long i = (long)blockIdx.x * blockDim.x + threadIdx.x;
    if (i < n) { nmax[i] = orderF(-INFINITY); denom[i] = 0.f; }
}

// ---------------- edge pass 1: segment max of leaky-relu scores ----------------
__global__ __launch_bounds__(256) void k_edge_max(const int* __restrict__ ei,
                                                  const float* __restrict__ a_s,
                                                  const float* __restrict__ a_d,
                                                  int* __restrict__ nmax, int E, int EE) {
    long t = (long)blockIdx.x * blockDim.x + threadIdx.x;
    if (t >= (long)EE * 4) return;
    int eid = (int)(t >> 2), hd = (int)(t & 3);
    int src, dst;
    if (eid < E) { src = ei[eid]; dst = ei[E + eid]; } else { src = dst = eid - E; }
    float e = a_s[src * 4 + hd] + a_d[dst * 4 + hd];
    e = (e > 0.f) ? e : NEG_SLOPE * e;
    atomicMax(&nmax[dst * 4 + hd], orderF(e));
}

// ---------------- edge pass 2: e_exp + segment-sum denominator ----------------
__global__ __launch_bounds__(256) void k_edge_exp(const int* __restrict__ ei,
                                                  const float* __restrict__ a_s,
                                                  const float* __restrict__ a_d,
                                                  const int* __restrict__ nmax,
                                                  float* __restrict__ eexp,
                                                  float* __restrict__ denom, int E, int EE) {
    long t = (long)blockIdx.x * blockDim.x + threadIdx.x;
    if (t >= (long)EE * 4) return;
    int eid = (int)(t >> 2), hd = (int)(t & 3);
    int src, dst;
    if (eid < E) { src = ei[eid]; dst = ei[E + eid]; } else { src = dst = eid - E; }
    float e = a_s[src * 4 + hd] + a_d[dst * 4 + hd];
    e = (e > 0.f) ? e : NEG_SLOPE * e;
    float ex = __expf(e - unorderF(nmax[dst * 4 + hd]));
    eexp[(size_t)eid * 4 + hd] = ex;
    atomicAdd(&denom[dst * 4 + hd], ex);
}

// ---------------- edge pass 3: agg[dst] += alpha * h[src] (memory-bound core) ------
// 4 edges per block; 64 threads per edge; each thread moves a float4 (b128 loads).
__global__ __launch_bounds__(256) void k_edge_agg(const int* __restrict__ ei,
                                                  const float* __restrict__ hlin,
                                                  const float* __restrict__ eexp,
                                                  const float* __restrict__ denom,
                                                  float* __restrict__ agg, int E, int EE) {
    int eid = blockIdx.x * 4 + (threadIdx.x >> 6);  // wave-uniform -> scalar index loads
    if (eid >= EE) return;
    int src, dst;
    if (eid < E) { src = ei[eid]; dst = ei[E + eid]; } else { src = dst = eid - E; }
    int j  = threadIdx.x & 63;   // float4 group 0..63 (channel j*4..j*4+3)
    int hd = j >> 4;
    float alpha = eexp[(size_t)eid * 4 + hd] / (denom[(size_t)dst * 4 + hd] + EPSV);
    const float4 v = *(const float4*)(hlin + (size_t)src * 256 + (size_t)j * 4);
    float* o = agg + (size_t)dst * 256 + (size_t)j * 4;
    atomicAdd(o + 0, alpha * v.x);
    atomicAdd(o + 1, alpha * v.y);
    atomicAdd(o + 2, alpha * v.z);
    atomicAdd(o + 3, alpha * v.w);
}

// ---------------- relu (+ optional f16 re-convert for next layer) ----------------
__global__ __launch_bounds__(256) void k_relu_cvt(float* __restrict__ x,
                                                  _Float16* __restrict__ y, long n) {
    long i = (long)blockIdx.x * blockDim.x + threadIdx.x;
    if (i >= n) return;
    float v = x[i];
    v = v > 0.f ? v : 0.f;
    x[i] = v;
    if (y) y[i] = (_Float16)v;
}

// ---------------- readout: column mean then tiny GEMV ----------------
__global__ void k_zero(float* __restrict__ p, int n) {
    int i = blockIdx.x * blockDim.x + threadIdx.x;
    if (i < n) p[i] = 0.f;
}
__global__ __launch_bounds__(256) void k_colsum(const float* __restrict__ h,
                                                float* __restrict__ emb, int N) {
    int rpb = (N + gridDim.x - 1) / gridDim.x;
    int r0 = blockIdx.x * rpb;
    int r1 = min(r0 + rpb, N);
    int c = threadIdx.x;
    float s = 0.f;
    for (int r = r0; r < r1; ++r) s += h[(size_t)r * 256 + c];
    atomicAdd(&emb[c], s);
}
__global__ __launch_bounds__(64) void k_head(const float* __restrict__ emb,
                                             const float* __restrict__ Wout,
                                             const float* __restrict__ bout,
                                             float* __restrict__ out, float invN) {
    int c = threadIdx.x;  // 0..63
    float s = 0.f;
    for (int hc = 0; hc < 256; ++hc) s += emb[hc] * invN * Wout[hc * 64 + c];
    out[c] = s + bout[c];
}

// =============================== host launcher ===============================
extern "C" void kernel_launch(void* const* d_in, const int* in_sizes, int n_in,
                              void* d_out, int out_size, void* d_ws, size_t ws_size,
                              hipStream_t stream) {
    const float* x    = (const float*)d_in[0];
    const int*   ei   = (const int*)d_in[1];     // edge_index [2,E]
    const float* W0   = (const float*)d_in[2];
    const float* as0  = (const float*)d_in[3];
    const float* ad0  = (const float*)d_in[4];
    const float* b0   = (const float*)d_in[5];
    const float* W1   = (const float*)d_in[6];
    const float* as1  = (const float*)d_in[7];
    const float* ad1  = (const float*)d_in[8];
    const float* b1   = (const float*)d_in[9];
    const float* Wout = (const float*)d_in[10];
    const float* bout = (const float*)d_in[11];
    float* out = (float*)d_out;

    const int HC  = in_sizes[5];              // 256
    const int Fin = in_sizes[2] / HC;         // 128
    const int N   = in_sizes[0] / Fin;        // 50000
    const int E   = in_sizes[1] / 2;          // 800000
    const int EE  = E + N;                    // edges + self loops

    // ---- workspace carve ----
    char* base = (char*)d_ws;
    size_t off = 0;
    auto carve = [&](size_t bytes) { void* p = base + off; off += (bytes + 255) & ~(size_t)255; return p; };
    _Float16* fh    = (_Float16*)carve((size_t)N * HC * sizeof(_Float16)); // f16 acts (layer in)
    _Float16* wt    = (_Float16*)carve((size_t)HC * HC * sizeof(_Float16)); // transposed weights
    float*    hlin  = (float*)carve((size_t)N * HC * sizeof(float));       // x@W (pre-bias)
    float*    agg   = (float*)carve((size_t)N * HC * sizeof(float));       // layer output
    float*    a_s   = (float*)carve((size_t)N * 4 * sizeof(float));
    float*    a_d   = (float*)carve((size_t)N * 4 * sizeof(float));
    int*      nmax  = (int*)carve((size_t)N * 4 * sizeof(int));
    float*    denom = (float*)carve((size_t)N * 4 * sizeof(float));
    float*    eexp  = (float*)carve((size_t)EE * 4 * sizeof(float));
    float*    emb   = (float*)carve(256 * sizeof(float));
    (void)ws_size; (void)n_in; (void)out_size;

    auto blocks = [](long n, int t) { return (unsigned)((n + t - 1) / t); };

    auto run_layer = [&](int K, const float* W, const float* att_s, const float* att_d,
                         const float* bias) {
        // weights: W[K,HC] -> wt[HC,K] f16
        k_transpose_f16<<<blocks((long)K * HC, 256), 256, 0, stream>>>(W, wt, K, HC);
        // GEMM: hlin = fh @ wt^T  (one wave per 16x64 strip, 4 WMMAs per K-step)
        long waves = (long)(N / 16) * (HC / 64);
        k_gemm_wmma<<<blocks(waves * 32, 256), 256, 0, stream>>>(fh, wt, hlin, N, K, HC);
        // attention scores
        k_attn<<<N, 128, 0, stream>>>(hlin, att_s, att_d, a_s, a_d, N);
        // softmax state + bias-initialized output
        k_init_softmax<<<blocks((long)N * 4, 256), 256, 0, stream>>>(nmax, denom, (long)N * 4);
        k_init_agg<<<blocks((long)N * HC, 256), 256, 0, stream>>>(bias, agg, (long)N * HC);
        // edge passes
        k_edge_max<<<blocks((long)EE * 4, 256), 256, 0, stream>>>(ei, a_s, a_d, nmax, E, EE);
        k_edge_exp<<<blocks((long)EE * 4, 256), 256, 0, stream>>>(ei, a_s, a_d, nmax, eexp, denom, E, EE);
        k_edge_agg<<<blocks((long)EE, 4) , 256, 0, stream>>>(ei, hlin, eexp, denom, agg, E, EE);
    };

    // layer 0: convert x -> f16
    k_cvt_f16<<<blocks((long)N * Fin, 256), 256, 0, stream>>>(x, fh, (long)N * Fin);
    run_layer(Fin, W0, as0, ad0, b0);
    // relu + convert to f16 for layer 1
    k_relu_cvt<<<blocks((long)N * HC, 256), 256, 0, stream>>>(agg, fh, (long)N * HC);
    // layer 1
    run_layer(HC, W1, as1, ad1, b1);
    // relu (no convert needed)
    k_relu_cvt<<<blocks((long)N * HC, 256), 256, 0, stream>>>(agg, (_Float16*)nullptr, (long)N * HC);
    // readout
    k_zero<<<1, 256, 0, stream>>>(emb, 256);
    k_colsum<<<512, 256, 0, stream>>>(agg, emb, N);
    k_head<<<1, 64, 0, stream>>>(emb, Wout, bout, out, 1.0f / (float)N);
}
